// GraphAttentionLayer_47158740910295
// MI455X (gfx1250) — matrix operand order repaired and checked by reading, not compiled, not run on hardware
//
#include <hip/hip_runtime.h>
#include <hip/hip_bf16.h>

#define ALPHA 0.2f

typedef _Float16 v16h __attribute__((ext_vector_type(16)));
typedef float    v8f  __attribute__((ext_vector_type(8)));

constexpr int NB = 8;     // batches
constexpr int NN = 2048;  // nodes
constexpr int NF = 64;    // features

__device__ __forceinline__ float lrelu(float x) { return x > 0.f ? x : ALPHA * x; }

// ---------------------------------------------------------------------------
// Kernel 1: h = inp @ W (fp32), store h transposed as f16 hT[b][col][row],
//           plus per-row f1 = h.a1, f2 = h.a2.  One thread per (b,row).
// ---------------------------------------------------------------------------
__global__ __launch_bounds__(256) void gat_prep(
    const float* __restrict__ inp, const float* __restrict__ W,
    const float* __restrict__ a,
    _Float16* __restrict__ hT, float* __restrict__ f1, float* __restrict__ f2)
{
  __shared__ float Ws[NF * NF];
  for (int t = threadIdx.x; t < NF * NF; t += blockDim.x) Ws[t] = W[t];
  __syncthreads();

  const int idx = blockIdx.x * blockDim.x + threadIdx.x;   // [0, 16384)
  const int b   = idx >> 11;
  const int row = idx & (NN - 1);

  const float* xr = inp + (size_t)idx * NF;
  float x[NF];
  #pragma unroll
  for (int k = 0; k < NF; k += 4) {
    float4 v = *(const float4*)(xr + k);
    x[k] = v.x; x[k + 1] = v.y; x[k + 2] = v.z; x[k + 3] = v.w;
  }

  float s1 = 0.f, s2 = 0.f;
  for (int c = 0; c < NF; ++c) {
    float hc = 0.f;
    #pragma unroll
    for (int k = 0; k < NF; ++k) hc = fmaf(x[k], Ws[k * NF + c], hc);
    s1 = fmaf(hc, a[c], s1);
    s2 = fmaf(hc, a[NF + c], s2);
    hT[((size_t)b * NF + c) * NN + row] = (_Float16)hc;   // coalesced across lanes
  }
  f1[idx] = s1;
  f2[idx] = s2;
}

// ---------------------------------------------------------------------------
// Kernel 2: per-batch max of f2 (for the softmax upper bound).
// ---------------------------------------------------------------------------
__global__ __launch_bounds__(256) void gat_rowmax(
    const float* __restrict__ f2, float* __restrict__ f2max)
{
  __shared__ float red[256];
  const int b = blockIdx.x;
  float m = -1e30f;
  for (int j = threadIdx.x; j < NN; j += 256) m = fmaxf(m, f2[b * NN + j]);
  red[threadIdx.x] = m;
  __syncthreads();
  for (int s = 128; s > 0; s >>= 1) {
    if (threadIdx.x < (unsigned)s)
      red[threadIdx.x] = fmaxf(red[threadIdx.x], red[threadIdx.x + s]);
    __syncthreads();
  }
  if (threadIdx.x == 0) f2max[b] = red[0];
}

// ---------------------------------------------------------------------------
// Kernel 3: fused masked-softmax attention + (attn @ h) + ELU.
// One wave32 per 16-row i-tile. j processed 32 at a time (one WMMA K-step).
// A-frag (16x32 f16): lane L<16 holds row L, K = 0..15 (contiguous);
//                     lane L+16 holds row L, K = 16..31.
// B-frag (32x16 f16): lane L<16 holds col L, K = 0..15 (from hT, contiguous);
//                     lane L+16 holds col L-16, K = 16..31.
// Row sums computed on the matrix pipe: accl = P @ ones(32x16); every column
// of that D tile equals the row sum, so lane L's VGPR r is l[r + hf*8] --
// exactly the row this lane writes out.  No shuffles, no scalar adds.
// ---------------------------------------------------------------------------
__global__ __launch_bounds__(32) void gat_attn(
    const int* __restrict__ adj, const _Float16* __restrict__ hT,
    const float* __restrict__ f1, const float* __restrict__ f2,
    const float* __restrict__ f2max, float* __restrict__ out)
{
  const int tile = blockIdx.x;           // [0, 1024)
  const int b    = tile >> 7;            // 128 i-tiles per batch
  const int i0   = (tile & 127) << 4;
  const int lane = threadIdx.x;
  const int m    = lane & 15;
  const int hf   = lane >> 4;

  const float f1v = f1[b * NN + i0 + m];
  const float Mi  = lrelu(f1v + f2max[b]);         // per-row safe upper bound

  const int*      adjr = adj + (size_t)(b * NN + i0 + m) * NN;
  const float*    f2b  = f2 + b * NN;
  const _Float16* hTb  = hT + (size_t)b * NF * NN;

  v16h ones;
  #pragma unroll
  for (int t = 0; t < 16; ++t) ones[t] = (_Float16)1.f;

  v8f acc[4] = {v8f{}, v8f{}, v8f{}, v8f{}};
  v8f accl = {};                                   // row-sum tile

  for (int j0 = 0; j0 < NN; j0 += 32) {
    __builtin_prefetch((const void*)(adjr + j0 + 32), 0, 0);  // global_prefetch_b8
                                                              // (speculative: safe past end)
    // j indices covered by this lane: jb0+e (e=0..7), jb1+(e-8) (e=8..15), i.e. j0+16*hf+e
    const int jb0 = j0 + hf * 16;
    const int jb1 = j0 + hf * 16 + 8;

    int   aj[16] __attribute__((aligned(16)));
    float fj[16] __attribute__((aligned(16)));
    ((int4*)aj)[0] = *(const int4*)(adjr + jb0);
    ((int4*)aj)[1] = *(const int4*)(adjr + jb0 + 4);
    ((int4*)aj)[2] = *(const int4*)(adjr + jb1);
    ((int4*)aj)[3] = *(const int4*)(adjr + jb1 + 4);
    ((float4*)fj)[0] = *(const float4*)(f2b + jb0);
    ((float4*)fj)[1] = *(const float4*)(f2b + jb0 + 4);
    ((float4*)fj)[2] = *(const float4*)(f2b + jb1);
    ((float4*)fj)[3] = *(const float4*)(f2b + jb1 + 4);

    // Issue all four B-fragment loads up front so the 8x b128 clause overlaps
    // with the exp/mask VALU work below.
    v16h hb[4];
    #pragma unroll
    for (int n = 0; n < 4; ++n) {
      const _Float16* hp = hTb + (size_t)(n * 16 + m) * NN + (j0 + hf * 16);
      hb[n] = *(const v16h*)hp;                     // 32B contiguous load
    }

    // Branchless P fragment: unconditional exp, mask folded in as a multiply.
    v16h pa;
    #pragma unroll
    for (int e = 0; e < 16; ++e) {
      const float s    = f1v + fj[e];
      const float el   = (s > 0.f ? s : ALPHA * s) - Mi;   // v_cndmask, no branch
      const float mask = aj[e] > 0 ? 1.f : 0.f;
      pa[e] = (_Float16)(__expf(el) * mask);
    }

    #pragma unroll
    for (int n = 0; n < 4; ++n)
      acc[n] = __builtin_amdgcn_wmma_f32_16x16x32_f16(
          false, pa, false, hb[n], (short)0, acc[n], false, false);
    // Row sums on the matrix pipe.
    accl = __builtin_amdgcn_wmma_f32_16x16x32_f16(
        false, pa, false, ones, (short)0, accl, false, false);
  }

  // D layout: VGPR r -> row r (lane<16) / row r+8 (lane>=16), col = n*16 + m.
  float* outb = out + (size_t)(b * NN + i0) * NF;
  #pragma unroll
  for (int r = 0; r < 8; ++r) {
    const int row = r + hf * 8;
    const float inv = __builtin_amdgcn_rcpf(accl[r]);  // v_rcp_f32, no IEEE div chain
    #pragma unroll
    for (int n = 0; n < 4; ++n) {
      const float v = acc[n][r] * inv;
      // branchless ELU: max(v,0) + exp(min(v,0)) - 1  (exact)
      outb[(size_t)row * NF + n * 16 + m] =
          fmaxf(v, 0.f) + __expf(fminf(v, 0.f)) - 1.f;
    }
  }
}

// ---------------------------------------------------------------------------
extern "C" void kernel_launch(void* const* d_in, const int* in_sizes, int n_in,
                              void* d_out, int out_size, void* d_ws, size_t ws_size,
                              hipStream_t stream) {
  const float* inp = (const float*)d_in[0];   // (8, 2048, 64) f32
  const int*   adj = (const int*)d_in[1];     // (8, 2048, 2048) i32
  const float* W   = (const float*)d_in[2];   // (64, 64) f32
  const float* a   = (const float*)d_in[3];   // (128, 1) f32
  float* out = (float*)d_out;                 // (8, 2048, 64) f32

  char* ws = (char*)d_ws;
  _Float16* hT   = (_Float16*)ws;                               // 2 MB
  float*    f1   = (float*)(ws + 2u * 1024 * 1024);             // 64 KB
  float*    f2   = (float*)(ws + 2u * 1024 * 1024 + 64 * 1024); // 64 KB
  float*    f2mx = (float*)(ws + 2u * 1024 * 1024 + 128 * 1024);

  gat_prep<<<NB * NN / 256, 256, 0, stream>>>(inp, W, a, hT, f1, f2);
  gat_rowmax<<<NB, 256, 0, stream>>>(f2, f2mx);
  gat_attn<<<NB * (NN / 16), 32, 0, stream>>>(adj, hT, f1, f2, f2mx, out);
}
